// FixedGateRNN_3968549782189
// MI455X (gfx1250) — matrix operand order repaired and checked by reading, not compiled, not run on hardware
//
#include <hip/hip_runtime.h>

#define DIM    512
#define VOCABN 32000
#define BATCH  32
#define SEQ    4096
#define TK     1024   // 0.9^k underflows f32 past k~983; earlier t contribute exactly 0
#define NT     8      // t-slices per (b, col-group) for parallelism
#define LOGG   (-0.10536051565782628f)  // ln(0.9)
#define OMG    0.1f                     // 1 - G

#define NCT    (DIM / 16)     // 32 column tiles in stage A
#define NKC    (DIM / 32)     // 16 K-chunks of 32
#define PACK_LANES (NCT * NKC * 32)          // 16384 lane-segments
#define PACK_ELEMS (PACK_LANES * 16)         // 262144 f16 = 512 KB

typedef __attribute__((ext_vector_type(16))) _Float16 v16h;
typedef __attribute__((ext_vector_type(8)))  float    v8f;

// ---- fragment loaders ------------------------------------------------------

// B-fragment from pre-packed f16: lane's 16 values are 32 contiguous bytes.
__device__ __forceinline__ v16h load_b_frag_packed(const _Float16* __restrict__ p) {
    union { uint4 q[2]; v16h h; } u;
    u.q[0] = *(const uint4*)(p);
    u.q[1] = *(const uint4*)(p + 8);
    return u.h;
}

// B-fragment (32x16 f16, K x N) direct from f32: lane holds column n = lane%16;
// lanes 0-15 hold K=0..15, lanes 16-31 hold K=16..31 (pairs ascending).
__device__ __forceinline__ v16h load_b_frag_f32(const float* __restrict__ p) {
    float4 q0 = *(const float4*)(p);
    float4 q1 = *(const float4*)(p + 4);
    float4 q2 = *(const float4*)(p + 8);
    float4 q3 = *(const float4*)(p + 12);
    v16h r;
    r[0]  = (_Float16)q0.x; r[1]  = (_Float16)q0.y; r[2]  = (_Float16)q0.z; r[3]  = (_Float16)q0.w;
    r[4]  = (_Float16)q1.x; r[5]  = (_Float16)q1.y; r[6]  = (_Float16)q1.z; r[7]  = (_Float16)q1.w;
    r[8]  = (_Float16)q2.x; r[9]  = (_Float16)q2.y; r[10] = (_Float16)q2.z; r[11] = (_Float16)q2.w;
    r[12] = (_Float16)q3.x; r[13] = (_Float16)q3.y; r[14] = (_Float16)q3.z; r[15] = (_Float16)q3.w;
    return r;
}

// A-fragment (16x32 f16, M x K): lane holds row m = lane%16.
// Lanes 0-15: K=0..15; lanes 16-31: K=16..31 (contiguous, matches B).
__device__ __forceinline__ v16h load_a_frag(const float* __restrict__ rowptr,
                                            int kk, int lane) {
    int base = (lane < 16) ? 0 : 16;
    const float* p = rowptr + kk + base;
    float4 q0 = *(const float4*)(p);
    float4 q1 = *(const float4*)(p + 4);
    float4 q2 = *(const float4*)(p + 8);
    float4 q3 = *(const float4*)(p + 12);
    v16h r;
    r[0]  = (_Float16)q0.x; r[1]  = (_Float16)q0.y; r[2]  = (_Float16)q0.z; r[3]  = (_Float16)q0.w;
    r[4]  = (_Float16)q1.x; r[5]  = (_Float16)q1.y; r[6]  = (_Float16)q1.z; r[7]  = (_Float16)q1.w;
    r[8]  = (_Float16)q2.x; r[9]  = (_Float16)q2.y; r[10] = (_Float16)q2.z; r[11] = (_Float16)q2.w;
    r[12] = (_Float16)q3.x; r[13] = (_Float16)q3.y; r[14] = (_Float16)q3.z; r[15] = (_Float16)q3.w;
    return r;
}

// ---- prep kernels ----------------------------------------------------------

__global__ void fixedgate_zero_kernel(float* __restrict__ p, int n) {
    int i = blockIdx.x * blockDim.x + threadIdx.x;
    if (i < n) p[i] = 0.0f;
}

// Pack W_w (f32, [e, d]) into per-lane B-fragment segments (f16).
// pack[((ct*NKC + kc)*32 + lane)*16 + j] = W_w[ct*16 + lane%16, kc*32 + koff + j]
__global__ void fixedgate_pack_w_kernel(const float* __restrict__ Ww,
                                        _Float16* __restrict__ pack) {
    int tid = blockIdx.x * blockDim.x + threadIdx.x;     // 0..PACK_LANES-1
    if (tid >= PACK_LANES) return;
    int lane = tid & 31;
    int kc   = (tid >> 5) & (NKC - 1);
    int ct   = tid >> 9;
    int n    = lane & 15;
    int koff = (lane < 16) ? 0 : 16;
    const float* src = Ww + (size_t)(ct * 16 + n) * DIM + kc * 32 + koff;
    _Float16* dst = pack + (size_t)tid * 16;
#pragma unroll
    for (int j = 0; j < 16; ++j) dst[j] = (_Float16)src[j];
}

// ---- Stage A: h[b,e] += sum_t w_t * tanh(emb[x[b,t]] . W_w[e,:] + W_b[e]) --
// One wave per (col-pair, b, t-slice): 16x32 output tile, A fragment shared
// by 2 WMMAs, B operands pre-packed f16 (no in-loop conversion).
__global__ __launch_bounds__(32)
void fixedgate_rnn_accum_kernel(const int*      __restrict__ x,
                                const float*    __restrict__ emb,
                                const _Float16* __restrict__ packW,
                                const float*    __restrict__ Wb,
                                float*          __restrict__ h) {
    const int lane   = threadIdx.x;
    const int ct0    = blockIdx.x * 2;            // column tiles ct0, ct0+1
    const int b      = blockIdx.y;
    const int zslice = blockIdx.z;

    const int n     = lane & 15;
    const int e0    = ct0 * 16 + n;
    const int e1    = e0 + 16;
    const float bias0 = Wb[e0];
    const float bias1 = Wb[e1];
    const int mbase = (lane < 16) ? 0 : 8;        // C-fragment row base

    const _Float16* pw0 = packW + ((size_t)(ct0      * NKC) * 32 + lane) * 16;
    const _Float16* pw1 = packW + ((size_t)((ct0 + 1) * NKC) * 32 + lane) * 16;

    const int tstart = SEQ - TK + zslice * (TK / NT);
    float acc0 = 0.0f, acc1 = 0.0f;

    for (int chunk = 0; chunk < (TK / NT) / 16; ++chunk) {
        const int t0  = tstart + chunk * 16;
        const int tok = x[b * SEQ + t0 + (lane & 15)];    // A row m = lane%16
        const float* arow = emb + (size_t)tok * DIM;

        v8f c0 = {}, c1 = {};
#pragma unroll
        for (int kc = 0; kc < NKC; ++kc) {
            v16h a  = load_a_frag(arow, kc * 32, lane);
            v16h w0 = load_b_frag_packed(pw0 + (size_t)kc * 32 * 16);
            v16h w1 = load_b_frag_packed(pw1 + (size_t)kc * 32 * 16);
            c0 = __builtin_amdgcn_wmma_f32_16x16x32_f16(
                     false, a, false, w0, (short)0, c0, false, false);
            c1 = __builtin_amdgcn_wmma_f32_16x16x32_f16(
                     false, a, false, w1, (short)0, c1, false, false);
        }
#pragma unroll
        for (int i = 0; i < 8; ++i) {
            const int t = t0 + mbase + i;                 // row of this C VGPR
            const float w = OMG * __expf((float)(SEQ - 1 - t) * LOGG);
            acc0 += w * tanhf(c0[i] + bias0);
            acc1 += w * tanhf(c1[i] + bias1);
        }
    }
    // lanes l and l+16 hold the same column n = l%16 (row halves 0-7 / 8-15)
    acc0 += __shfl_xor(acc0, 16, 32);
    acc1 += __shfl_xor(acc1, 16, 32);
    if (lane < 16) {
        atomicAdd(&h[b * DIM + e0], acc0);
        atomicAdd(&h[b * DIM + e1], acc1);
    }
}

// ---- Stage B: out[b,v] = h[b,:] . head_w[v,:] + head_b[v] ------------------
// One wave per 16x32 (b, v) tile: A fragment from h shared by 2 WMMAs;
// head_w streamed once, coalesced 64B/lane.
__global__ __launch_bounds__(32)
void fixedgate_head_kernel(const float* __restrict__ h,
                           const float* __restrict__ head_w,
                           const float* __restrict__ head_b,
                           float*       __restrict__ out) {
    const int lane  = threadIdx.x;
    const int v0    = blockIdx.x * 32;            // two vocab tiles: v0, v0+16
    const int bTile = blockIdx.y;                 // 0..1

    const int n  = lane & 15;
    const int va = v0 + n;
    const int vb = va + 16;
    const float* arow  = h + (size_t)(bTile * 16 + (lane & 15)) * DIM;
    const float* wrow0 = head_w + (size_t)va * DIM;
    const float* wrow1 = head_w + (size_t)vb * DIM;
    const int koff  = (lane < 16) ? 0 : 16;
    const int mbase = (lane < 16) ? 0 : 8;

    v8f c0 = {}, c1 = {};
#pragma unroll
    for (int kk = 0; kk < DIM; kk += 32) {
        v16h a  = load_a_frag(arow, kk, lane);
        v16h w0 = load_b_frag_f32(wrow0 + kk + koff);
        v16h w1 = load_b_frag_f32(wrow1 + kk + koff);
        c0 = __builtin_amdgcn_wmma_f32_16x16x32_f16(
                 false, a, false, w0, (short)0, c0, false, false);
        c1 = __builtin_amdgcn_wmma_f32_16x16x32_f16(
                 false, a, false, w1, (short)0, c1, false, false);
    }
    const float bias0 = head_b[va];
    const float bias1 = head_b[vb];
#pragma unroll
    for (int i = 0; i < 8; ++i) {
        const size_t row = (size_t)(bTile * 16 + mbase + i) * VOCABN;
        out[row + va] = c0[i] + bias0;
        out[row + vb] = c1[i] + bias1;
    }
}

// ---- launcher --------------------------------------------------------------

extern "C" void kernel_launch(void* const* d_in, const int* in_sizes, int n_in,
                              void* d_out, int out_size, void* d_ws, size_t ws_size,
                              hipStream_t stream) {
    const int*   x      = (const int*)  d_in[0];   // [32, 4096]
    const float* emb    = (const float*)d_in[1];   // [32000, 512]
    const float* Ww     = (const float*)d_in[2];   // [512, 512]
    const float* Wb     = (const float*)d_in[3];   // [512]
    const float* head_w = (const float*)d_in[4];   // [32000, 512]
    const float* head_b = (const float*)d_in[5];   // [32000]
    float* out = (float*)d_out;                    // [32, 32000]

    float*    h     = (float*)d_ws;                          // 64 KB
    _Float16* packW = (_Float16*)((char*)d_ws + (size_t)BATCH * DIM * sizeof(float)); // 512 KB

    // h must start at zero (ws is poisoned by the harness)
    fixedgate_zero_kernel<<<(BATCH * DIM + 255) / 256, 256, 0, stream>>>(h, BATCH * DIM);

    // one-time f16 fragment pack of W_w (1 MB read, 512 KB write; L2-resident)
    fixedgate_pack_w_kernel<<<(PACK_LANES + 255) / 256, 256, 0, stream>>>(Ww, packW);

    dim3 gA(DIM / 32, BATCH, NT);                  // 16 x 32 x 8 = 4096 waves
    fixedgate_rnn_accum_kernel<<<gA, 32, 0, stream>>>(x, emb, packW, Wb, h);

    dim3 gB(VOCABN / 32, BATCH / 16);              // 1000 x 2 = 2000 waves
    fixedgate_head_kernel<<<gB, 32, 0, stream>>>(h, head_w, head_b, out);
}
